// GraphSAGE_13185549598985
// MI455X (gfx1250) — matrix operand order repaired and checked by reading, not compiled, not run on hardware
//
#include <hip/hip_runtime.h>

#define NNODES 200000
#define NEDGES 1600000
#define HIDDEN 128
#define BROWS  8192
#define EPSV   1e-5f

typedef __bf16 bf16;
typedef bf16  v16bf __attribute__((ext_vector_type(16)));
typedef bf16  v8bf  __attribute__((ext_vector_type(8)));
typedef float v8f   __attribute__((ext_vector_type(8)));

// Swizzled fragment layouts (this file's contiguous k order, same for A and B):
//  B (32x16 per chunk): lane = half*16 + (n&15), element i -> k = chunk*32 + half*16 + i.
//    Wsw index: ((tile*chunks + chunk)*32 + lane)*16 + i   -> per-lane contiguous 32B.
//  A (16x32 per chunk): lane = half*16 + m, element i -> k = chunk*32 + half*16 + i.
//    Each half-fragment (i in [j*8, j*8+8)) covers 8 consecutive k -> staged as one 16B store.
//    sA index: half-fragment f = chunk*64 + lane*2 + j occupies elements [f*8, f*8+8).

// ---------- Weight pre-swizzle (f32 [K,128] row-major -> bf16 fragment-major) ----------
__global__ void prep_weight_kernel(const float* __restrict__ W, bf16* __restrict__ Wsw, int K) {
  int t = blockIdx.x * blockDim.x + threadIdx.x;
  if (t >= K * HIDDEN) return;
  int k = t >> 7, n = t & 127;
  int chunks = K >> 5;
  int tile  = n >> 4;
  int chunk = k >> 5;
  int hh    = (k >> 4) & 1;
  int i     = k & 15;
  int lane  = hh * 16 + (n & 15);
  Wsw[(((tile * chunks + chunk) * 32 + lane) << 4) + i] = (bf16)W[t];
}

// ---------- Utility kernels ----------
__global__ void zero_f32_kernel(float* __restrict__ p, long n) {
  long i = (long)blockIdx.x * blockDim.x + threadIdx.x;
  long stride = (long)gridDim.x * blockDim.x;
  for (; i < n; i += stride) p[i] = 0.f;
}

__global__ void deg_kernel(const int* __restrict__ row, float* __restrict__ deg, int e) {
  int i = blockIdx.x * blockDim.x + threadIdx.x;
  if (i < e) atomicAdd(&deg[row[i]], 1.0f);
}

__global__ void deg_inv_kernel(float* __restrict__ deg, int n) {
  int i = blockIdx.x * blockDim.x + threadIdx.x;
  if (i < n) deg[i] = 1.0f / fmaxf(deg[i], 1.0f);
}

__global__ void concat_kernel(const float* __restrict__ x,
                              const float* __restrict__ g0,
                              const float* __restrict__ g1,
                              const float* __restrict__ g2,
                              bf16* __restrict__ h, int n) {
  int i = blockIdx.x * blockDim.x + threadIdx.x;
  if (i >= n) return;
  bf16* r = h + (long)i * 32;
#pragma unroll
  for (int c = 0; c < 26; ++c) r[c] = (bf16)x[(long)i * 26 + c];
  r[26] = (bf16)g0[i * 2]; r[27] = (bf16)g0[i * 2 + 1];
  r[28] = (bf16)g1[i * 2]; r[29] = (bf16)g1[i * 2 + 1];
  r[30] = (bf16)g2[i * 2]; r[31] = (bf16)g2[i * 2 + 1];
}

// agg[row[e]] += h[col[e]]  (one 16B bf16 load, 8 f32 atomics per thread)
__global__ void edge_agg_kernel(const int* __restrict__ row, const int* __restrict__ col,
                                const bf16* __restrict__ h, float* __restrict__ agg,
                                int e, int C) {
  long t = (long)blockIdx.x * blockDim.x + threadIdx.x;
  const int chunks = C >> 3;
  long total = (long)e * chunks;
  if (t >= total) return;
  int eid = (int)(t / chunks);
  int c0  = (int)(t % chunks) << 3;
  v8bf hv = *(const v8bf*)(h + (long)col[eid] * C + c0);
  float* dst = agg + (long)row[eid] * C + c0;
#pragma unroll
  for (int j = 0; j < 8; ++j) atomicAdd(&dst[j], (float)hv[j]);
}

// ---------- Fused SAGE layer: y = (agg*deg_inv)@Wl + h@Wr + bl, + BN col stats ----------
__global__ __launch_bounds__(256) void sage_gemm_kernel(
    const float* __restrict__ agg, const float* __restrict__ deginv,
    const bf16* __restrict__ hin,
    const v16bf* __restrict__ Wlsw, const v16bf* __restrict__ Wrsw,
    const float* __restrict__ bias,
    float* __restrict__ y, float* __restrict__ bnsum, float* __restrict__ bnsumsq,
    int K) {
  __shared__ v16bf sA[128];          // (K/32)*32 fragments, 4 KB
  __shared__ v16bf sH[128];          // 4 KB
  __shared__ float sSum[HIDDEN];
  __shared__ float sSq[HIDDEN];

  const int tid  = threadIdx.x;
  const int lane = tid & 31;
  const int wave = tid >> 5;
  const int m0   = blockIdx.x * 16;
  const int chunks = K >> 5;

  if (tid < HIDDEN) { sSum[tid] = 0.f; sSq[tid] = 0.f; }

  // Stage A tiles pre-swizzled: half-fragment per thread, contiguous global reads.
  v8bf* sA8 = (v8bf*)sA;
  v8bf* sH8 = (v8bf*)sH;
  for (int f = tid; f < 2 * K; f += 256) {
    int j  = f & 1;
    int ln = (f >> 1) & 31;
    int ck = f >> 6;
    int m  = ln & 15, hh = ln >> 4;
    int kbase = ck * 32 + hh * 16 + j * 8;
    long r = m0 + m;
    const float4* ap = (const float4*)(agg + r * K + kbase);
    float4 a0 = ap[0], a1 = ap[1];
    float ds = deginv[r];
    v8bf av;
    av[0] = (bf16)(a0.x * ds); av[1] = (bf16)(a0.y * ds);
    av[2] = (bf16)(a0.z * ds); av[3] = (bf16)(a0.w * ds);
    av[4] = (bf16)(a1.x * ds); av[5] = (bf16)(a1.y * ds);
    av[6] = (bf16)(a1.z * ds); av[7] = (bf16)(a1.w * ds);
    sA8[f] = av;                                        // one ds_store_b128
    sH8[f] = *(const v8bf*)(hin + r * K + kbase);       // 16B copy
  }
  __syncthreads();

  v8f acc = {0.f, 0.f, 0.f, 0.f, 0.f, 0.f, 0.f, 0.f};
  // B fragments straight from pre-swizzled global weights (coalesced, WGP$-resident).
  for (int ck = 0; ck < chunks; ++ck) {
    v16bf a = sA[ck * 32 + lane];
    v16bf b = Wlsw[(wave * chunks + ck) * 32 + lane];
    acc = __builtin_amdgcn_wmma_f32_16x16x32_bf16(false, a, false, b, (short)0, acc, false, false);
  }
  for (int ck = 0; ck < chunks; ++ck) {
    v16bf a = sH[ck * 32 + lane];
    v16bf b = Wrsw[(wave * chunks + ck) * 32 + lane];
    acc = __builtin_amdgcn_wmma_f32_16x16x32_bf16(false, a, false, b, (short)0, acc, false, false);
  }

  // Epilogue: lane's 8 outputs share one column n -> fused BN stats.
  const int half = lane >> 4;
  const int n    = wave * 16 + (lane & 15);
  const float bv = bias[n];
  float s = 0.f, s2 = 0.f;
#pragma unroll
  for (int r = 0; r < 8; ++r) {
    int m = m0 + r + half * 8;
    float v = acc[r] + bv;
    y[(long)m * HIDDEN + n] = v;
    s += v; s2 += v * v;
  }
  atomicAdd(&sSum[n], s);
  atomicAdd(&sSq[n], s2);
  __syncthreads();
  if (tid < HIDDEN) {
    atomicAdd(&bnsum[tid],   sSum[tid]);
    atomicAdd(&bnsumsq[tid], sSq[tid]);
  }
}

// BN (training stats) + ReLU -> bf16 activations
__global__ void bn_relu_kernel(const float* __restrict__ y,
                               const float* __restrict__ bnsum, const float* __restrict__ bnsumsq,
                               const float* __restrict__ gamma, const float* __restrict__ beta,
                               bf16* __restrict__ hout, long total, float invM) {
  long i = (long)blockIdx.x * blockDim.x + threadIdx.x;
  long stride = (long)gridDim.x * blockDim.x;
  for (; i < total; i += stride) {
    int c = (int)(i & (HIDDEN - 1));
    float mean = bnsum[c] * invM;
    float var  = bnsumsq[c] * invM - mean * mean;
    float sc   = gamma[c] * rsqrtf(var + EPSV);
    float sh   = beta[c] - mean * sc;
    hout[i] = (bf16)fmaxf(y[i] * sc + sh, 0.f);
  }
}

// mlp1: z[b,i] = relu(hp[b,i,:8] @ w1 + b1) @ w2 + b2   (K=8: too skinny for 16x16x32 tiles)
__global__ void mlp1_kernel(const bf16* __restrict__ h, const int* __restrict__ po,
                            const float* __restrict__ w1, const float* __restrict__ b1,
                            const float* __restrict__ w2, const float* __restrict__ b2,
                            float* __restrict__ z) {
  int t = blockIdx.x * blockDim.x + threadIdx.x;
  if (t >= BROWS * HIDDEN) return;
  int b = t >> 7, i = t & 127;
  int node  = po[b * 8 + (i >> 4)];      // row-major reshape [65536,128]->[8192,128,8]
  int cbase = (i & 15) * 8;
  v8bf src = *(const v8bf*)(h + (long)node * HIDDEN + cbase);
  float hp[8];
#pragma unroll
  for (int j = 0; j < 8; ++j) hp[j] = (float)src[j];
  float acc = b2[0];
  for (int c = 0; c < HIDDEN; ++c) {
    float a = b1[c];
#pragma unroll
    for (int j = 0; j < 8; ++j) a += hp[j] * w1[j * HIDDEN + c];
    acc += fmaxf(a, 0.f) * w2[c];
  }
  z[t] = acc;
}

// per-column sum / sumsq over [rows,128]
__global__ void colstats_kernel(const float* __restrict__ a, long total,
                                float* __restrict__ sum, float* __restrict__ sumsq) {
  __shared__ float sS[HIDDEN], sQ[HIDDEN];
  int tid = threadIdx.x;
  if (tid < HIDDEN) { sS[tid] = 0.f; sQ[tid] = 0.f; }
  __syncthreads();
  long i = (long)blockIdx.x * blockDim.x + tid;
  long stride = (long)gridDim.x * blockDim.x;   // multiple of 128 -> column == tid&127
  int c = tid & 127;
  float ls = 0.f, lq = 0.f;
  for (; i < total; i += stride) { float v = a[i]; ls += v; lq += v * v; }
  atomicAdd(&sS[c], ls);
  atomicAdd(&sQ[c], lq);
  __syncthreads();
  if (tid < HIDDEN) { atomicAdd(&sum[tid], sS[tid]); atomicAdd(&sumsq[tid], sQ[tid]); }
}

// mlp2 head: A = relu(BN(z)) fused into swizzled A stage, t = relu(A@W1+b1) via WMMA,
// out = relu(t@W2+b2)
__global__ __launch_bounds__(256) void mlp2_kernel(
    const float* __restrict__ z,
    const float* __restrict__ zsum, const float* __restrict__ zsumsq,
    const float* __restrict__ gamma, const float* __restrict__ beta,
    const v16bf* __restrict__ W1sw, const float* __restrict__ b1,
    const float* __restrict__ W2, const float* __restrict__ b2,
    float* __restrict__ out) {
  __shared__ v16bf sA[128];
  __shared__ float sT[16 * HIDDEN];
  __shared__ float sScale[HIDDEN], sShift[HIDDEN];

  const int tid = threadIdx.x, lane = tid & 31, wave = tid >> 5;
  const int m0 = blockIdx.x * 16;
  const float invM = 1.0f / (float)BROWS;

  if (tid < HIDDEN) {
    float mean = zsum[tid] * invM;
    float var  = zsumsq[tid] * invM - mean * mean;
    float sc   = gamma[tid] * rsqrtf(var + EPSV);
    sScale[tid] = sc;
    sShift[tid] = beta[tid] - mean * sc;
  }
  __syncthreads();

  {  // stage A = relu(BN(z)) pre-swizzled, one half-fragment per thread
    int f  = tid;                 // 2*K = 256 half-fragments
    int j  = f & 1;
    int ln = (f >> 1) & 31;
    int ck = f >> 6;
    int m  = ln & 15, hh = ln >> 4;
    int kbase = ck * 32 + hh * 16 + j * 8;
    const float4* zp = (const float4*)(z + (long)(m0 + m) * HIDDEN + kbase);
    float4 z0 = zp[0], z1 = zp[1];
    float zz[8] = {z0.x, z0.y, z0.z, z0.w, z1.x, z1.y, z1.z, z1.w};
    v8bf av;
#pragma unroll
    for (int q = 0; q < 8; ++q)
      av[q] = (bf16)fmaxf(zz[q] * sScale[kbase + q] + sShift[kbase + q], 0.f);
    ((v8bf*)sA)[f] = av;
  }
  __syncthreads();

  v8f acc = {0.f, 0.f, 0.f, 0.f, 0.f, 0.f, 0.f, 0.f};
  for (int ck = 0; ck < 4; ++ck) {
    v16bf a = sA[ck * 32 + lane];
    v16bf b = W1sw[(wave * 4 + ck) * 32 + lane];
    acc = __builtin_amdgcn_wmma_f32_16x16x32_bf16(false, a, false, b, (short)0, acc, false, false);
  }

  const int half = lane >> 4;
  const int n    = wave * 16 + (lane & 15);
  const float bv = b1[n];
#pragma unroll
  for (int r = 0; r < 8; ++r) {
    int m = r + half * 8;
    sT[m * HIDDEN + n] = fmaxf(acc[r] + bv, 0.f);
  }
  __syncthreads();
  if (tid < 32) {  // 16 rows x 2 outputs
    int m = tid >> 1, o = tid & 1;
    float a = b2[o];
    for (int c = 0; c < HIDDEN; ++c) a += sT[m * HIDDEN + c] * W2[c * 2 + o];
    out[(long)(m0 + m) * 2 + o] = fmaxf(a, 0.f);
  }
}

// ---------- Driver ----------

extern "C" void kernel_launch(void* const* d_in, const int* in_sizes, int n_in,
                              void* d_out, int out_size, void* d_ws, size_t ws_size,
                              hipStream_t stream) {
  (void)in_sizes; (void)n_in; (void)out_size; (void)ws_size;
  const float* x    = (const float*)d_in[0];
  const float* g0   = (const float*)d_in[1];
  const float* g1   = (const float*)d_in[2];
  const float* g2   = (const float*)d_in[3];
  const int*   ei   = (const int*)d_in[4];
  const int*   po   = (const int*)d_in[5];
  const float* c0Wl = (const float*)d_in[6];
  const float* c0bl = (const float*)d_in[7];
  const float* c0Wr = (const float*)d_in[8];
  const float* cWl  = (const float*)d_in[9];
  const float* cbl  = (const float*)d_in[10];
  const float* cWr  = (const float*)d_in[11];
  const float* bng  = (const float*)d_in[12];
  const float* bnb  = (const float*)d_in[13];
  const float* m1w1 = (const float*)d_in[14];
  const float* m1b1 = (const float*)d_in[15];
  const float* m1w2 = (const float*)d_in[16];
  const float* m1b2 = (const float*)d_in[17];
  const float* bn2g = (const float*)d_in[18];
  const float* bn2b = (const float*)d_in[19];
  const float* m2w1 = (const float*)d_in[20];
  const float* m2b1 = (const float*)d_in[21];
  const float* m2w2 = (const float*)d_in[22];
  const float* m2b2 = (const float*)d_in[23];

  const int* row = ei;            // edge_index[0]
  const int* col = ei + NEDGES;   // edge_index[1]

  char* ws = (char*)d_ws;
  size_t off = 0;
  auto carve = [&](size_t bytes) -> void* {
    void* p = ws + off;
    off += (bytes + 255) & ~(size_t)255;
    return p;
  };
  float* agg  = (float*)carve((size_t)NNODES * HIDDEN * 4);   // 102.4 MB
  float* y    = (float*)carve((size_t)NNODES * HIDDEN * 4);   // 102.4 MB
  bf16*  h    = (bf16*) carve((size_t)NNODES * HIDDEN * 2);   //  51.2 MB
  float* deg  = (float*)carve((size_t)NNODES * 4);
  float* bns  = (float*)carve(HIDDEN * 4);
  float* bnq  = (float*)carve(HIDDEN * 4);
  float* z    = (float*)carve((size_t)BROWS * HIDDEN * 4);
  float* zs   = (float*)carve(HIDDEN * 4);
  float* zq   = (float*)carve(HIDDEN * 4);
  bf16* w0l_sw = (bf16*)carve((size_t)32 * HIDDEN * 2);
  bf16* w0r_sw = (bf16*)carve((size_t)32 * HIDDEN * 2);
  bf16* wl_sw  = (bf16*)carve((size_t)3 * HIDDEN * HIDDEN * 2);
  bf16* wr_sw  = (bf16*)carve((size_t)3 * HIDDEN * HIDDEN * 2);
  bf16* w21_sw = (bf16*)carve((size_t)HIDDEN * HIDDEN * 2);

  // ---- weight pre-swizzle (tiny; amortized over all GEMM blocks) ----
  prep_weight_kernel<<<16, 256, 0, stream>>>(c0Wl, w0l_sw, 32);
  prep_weight_kernel<<<16, 256, 0, stream>>>(c0Wr, w0r_sw, 32);
  for (int l = 0; l < 3; ++l) {
    prep_weight_kernel<<<64, 256, 0, stream>>>(cWl + (size_t)l * HIDDEN * HIDDEN,
                                               wl_sw + (size_t)l * HIDDEN * HIDDEN, HIDDEN);
    prep_weight_kernel<<<64, 256, 0, stream>>>(cWr + (size_t)l * HIDDEN * HIDDEN,
                                               wr_sw + (size_t)l * HIDDEN * HIDDEN, HIDDEN);
  }
  prep_weight_kernel<<<64, 256, 0, stream>>>(m2w1, w21_sw, HIDDEN);

  // ---- degree -> deg_inv (in place) ----
  zero_f32_kernel<<<2048, 256, 0, stream>>>(deg, NNODES);
  deg_kernel<<<(NEDGES + 255) / 256, 256, 0, stream>>>(row, deg, NEDGES);
  deg_inv_kernel<<<(NNODES + 255) / 256, 256, 0, stream>>>(deg, NNODES);

  // ---- layer 0 (K = 32) ----
  concat_kernel<<<(NNODES + 255) / 256, 256, 0, stream>>>(x, g0, g1, g2, h, NNODES);
  zero_f32_kernel<<<2048, 256, 0, stream>>>(agg, (long)NNODES * 32);
  {
    long t = (long)NEDGES * (32 / 8);
    edge_agg_kernel<<<(int)((t + 255) / 256), 256, 0, stream>>>(row, col, h, agg, NEDGES, 32);
  }
  zero_f32_kernel<<<1, 256, 0, stream>>>(bns, HIDDEN);
  zero_f32_kernel<<<1, 256, 0, stream>>>(bnq, HIDDEN);
  sage_gemm_kernel<<<NNODES / 16, 256, 0, stream>>>(agg, deg, h,
                                                    (const v16bf*)w0l_sw, (const v16bf*)w0r_sw,
                                                    c0bl, y, bns, bnq, 32);
  bn_relu_kernel<<<4096, 256, 0, stream>>>(y, bns, bnq, bng, bnb, h,
                                           (long)NNODES * HIDDEN, 1.0f / (float)NNODES);

  // ---- layers 1..3 (K = 128) ----
  for (int l = 0; l < 3; ++l) {
    zero_f32_kernel<<<4096, 256, 0, stream>>>(agg, (long)NNODES * HIDDEN);
    long t = (long)NEDGES * (HIDDEN / 8);
    edge_agg_kernel<<<(int)((t + 255) / 256), 256, 0, stream>>>(row, col, h, agg, NEDGES, HIDDEN);
    zero_f32_kernel<<<1, 256, 0, stream>>>(bns, HIDDEN);
    zero_f32_kernel<<<1, 256, 0, stream>>>(bnq, HIDDEN);
    sage_gemm_kernel<<<NNODES / 16, 256, 0, stream>>>(
        agg, deg, h,
        (const v16bf*)(wl_sw + (size_t)l * HIDDEN * HIDDEN),
        (const v16bf*)(wr_sw + (size_t)l * HIDDEN * HIDDEN),
        cbl + (size_t)l * HIDDEN, y, bns, bnq, HIDDEN);
    bn_relu_kernel<<<4096, 256, 0, stream>>>(y, bns, bnq,
                                             bng + (size_t)(l + 1) * HIDDEN,
                                             bnb + (size_t)(l + 1) * HIDDEN,
                                             h, (long)NNODES * HIDDEN, 1.0f / (float)NNODES);
  }

  // ---- head ----
  mlp1_kernel<<<(BROWS * HIDDEN + 255) / 256, 256, 0, stream>>>(h, po, m1w1, m1b1, m1w2, m1b2, z);
  zero_f32_kernel<<<1, 256, 0, stream>>>(zs, HIDDEN);
  zero_f32_kernel<<<1, 256, 0, stream>>>(zq, HIDDEN);
  colstats_kernel<<<256, 256, 0, stream>>>(z, (long)BROWS * HIDDEN, zs, zq);
  mlp2_kernel<<<BROWS / 16, 256, 0, stream>>>(z, zs, zq, bn2g, bn2b,
                                              (const v16bf*)w21_sw, m2b1, m2w2, m2b2,
                                              (float*)d_out);
}
